// MB_dMGC_CWTFFNet_57191784514134
// MI455X (gfx1250) — matrix operand order repaired and checked by reading, not compiled, not run on hardware
//
#include <hip/hip_runtime.h>

// ---- problem constants ----
#define BB   512
#define CCH  64          // EEG channels C
#define DD   512
#define DKW  64
#define DV3W 192
#define HIDW 128

typedef __attribute__((ext_vector_type(16))) __bf16 v16bf;
typedef __attribute__((ext_vector_type(8)))  __bf16 bf16x8;
typedef __attribute__((ext_vector_type(8)))  float  v8f;
typedef __attribute__((ext_vector_type(4)))  float  f4;

__device__ __forceinline__ __bf16 f2bf(float f) {
  union { float f; unsigned u; } c; c.f = f;
  unsigned r = c.u + 0x7FFFu + ((c.u >> 16) & 1u);   // round-to-nearest-even
  union { unsigned short s; __bf16 b; } o; o.s = (unsigned short)(r >> 16);
  return o.b;
}

__device__ __forceinline__ float eluf(float v) { return v > 0.f ? v : (__expf(v) - 1.f); }

// A fragment: 16x32 bf16 tile (row-major, leading dim ld).
// lanes 0-15: row=lane, K {0..7, 16..23}; lanes 16-31: row=lane-16, K {8..15, 24..31}
__device__ __forceinline__ v16bf ldA16(const __bf16* t, int ld, int lane) {
  int r  = lane & 15;
  int kb = (lane & 16) ? 8 : 0;
  const __bf16* p = t + r * ld + kb;
  bf16x8 lo = *(const bf16x8*)p;
  bf16x8 hi = *(const bf16x8*)(p + 16);
  v16bf o;
#pragma unroll
  for (int i = 0; i < 8; ++i) { o[i] = lo[i]; o[i + 8] = hi[i]; }
  return o;
}

// B fragment: 32x16 bf16, staged N-major (Bt[n][k], leading dim ld).
// lanes 0-15: col=lane, K {0..7, 16..23}; lanes 16-31: col=lane-16, K {8..15, 24..31}
__device__ __forceinline__ v16bf ldB16(const __bf16* t, int ld, int lane) {
  int n  = lane & 15;
  int kb = (lane & 16) ? 8 : 0;
  const __bf16* p = t + n * ld + kb;
  bf16x8 lo = *(const bf16x8*)p;
  bf16x8 hi = *(const bf16x8*)(p + 16);
  v16bf o;
#pragma unroll
  for (int i = 0; i < 8; ++i) { o[i] = lo[i]; o[i + 8] = hi[i]; }
  return o;
}

__device__ __forceinline__ v8f wmma_bf(v16bf a, v16bf b, v8f c) {
  return __builtin_amdgcn_wmma_f32_16x16x32_bf16(false, a, false, b, (short)0, c, false, false);
}

// =====================================================================
// Generic bf16 WMMA GEMM: Out = act(A[M,K] @ W[K,N] + bias (+ Cin))
// Block tile 128xBN (BN = 64 or 128), BK=32, 256 threads = 8 wave32.
//   BN=64 : wave owns 16x64 strip (1 A frag + 4 B frags -> 4 WMMA / kstep)
//   BN=128: wave owns 32x64 strip (2 A frags + 4 B frags -> 8 WMMA / kstep)
// Requires M%128==0, N%BN==0, K%32==0.  ACT: 0=none 1=relu 2=elu
// Prefetches next K-tile cachelines (global_prefetch_b8).
// =====================================================================
template <int ACT, bool HASC, int BN>
__global__ __launch_bounds__(256) void gemm_k(
    const float* __restrict__ A, const float* __restrict__ W,
    const float* __restrict__ bias, const float* __restrict__ Cin,
    float* __restrict__ Out, int M, int N, int K,
    long aS, long wS, long bS, long oS)
{
  constexpr int MW  = (BN == 128) ? 2 : 1;   // 16-row A frags per wave
  constexpr int BCH = BN / 32;               // f4 B chunks per thread per kstep
  __shared__ __align__(16) __bf16 sA[128 * 32];
  __shared__ __align__(16) __bf16 sB[BN * 32];   // N-major: sB[n*32+k]
  const int br = blockIdx.z;
  A    += (long)br * aS;
  W    += (long)br * wS;
  bias += (long)br * bS;
  Out  += (long)br * oS;
  const int t = threadIdx.x;
  const int wv = t >> 5, lane = t & 31;
  const int m0 = blockIdx.y * 128, n0 = blockIdx.x * BN;
  const int wr = (BN == 128) ? (wv & 3) : wv;   // wave row group
  const int wc = (BN == 128) ? (wv >> 2) : 0;   // wave col group (x64)

  const v8f vz = {0.f, 0.f, 0.f, 0.f, 0.f, 0.f, 0.f, 0.f};
  v8f acc[MW][4];
#pragma unroll
  for (int mi = 0; mi < MW; ++mi)
#pragma unroll
    for (int nt = 0; nt < 4; ++nt) acc[mi][nt] = vz;

  for (int k0 = 0; k0 < K; k0 += 32) {
    // stage A tile 128x32 (1024 f4 chunks, 4 per thread)
#pragma unroll
    for (int j = 0; j < 4; ++j) {
      int idx = t + j * 256;
      int r = idx >> 3, c = (idx & 7) << 2;
      f4 v = *(const f4*)(A + (long)(m0 + r) * K + k0 + c);
#pragma unroll
      for (int q = 0; q < 4; ++q) sA[r * 32 + c + q] = f2bf(v[q]);
    }
    // stage W tile 32xBN, transposed to N-major
#pragma unroll
    for (int j = 0; j < BCH; ++j) {
      int idx = t + j * 256;
      int kr = idx / (BN / 4), c = (idx % (BN / 4)) << 2;
      f4 v = *(const f4*)(W + (long)(k0 + kr) * N + n0 + c);
#pragma unroll
      for (int q = 0; q < 4; ++q) sB[(c + q) * 32 + kr] = f2bf(v[q]);
    }
    // prefetch next K-tile (one 128B line per thread) into caches
    if (k0 + 32 < K) {
      if (t < 128) {
        __builtin_prefetch(A + (long)(m0 + t) * K + k0 + 32, 0, 0);
      } else {
        int i2 = t - 128;
        int kr = i2 / BCH, c = (i2 % BCH) << 5;
        if (kr < 32) __builtin_prefetch(W + (long)(k0 + 32 + kr) * N + n0 + c, 0, 0);
      }
    }
    __syncthreads();
    v16bf af[MW];
#pragma unroll
    for (int mi = 0; mi < MW; ++mi)
      af[mi] = ldA16(sA + (wr * MW + mi) * 16 * 32, 32, lane);
#pragma unroll
    for (int nt = 0; nt < 4; ++nt) {
      v16bf bf_ = ldB16(sB + (wc * 64 + nt * 16) * 32, 32, lane);
#pragma unroll
      for (int mi = 0; mi < MW; ++mi)
        acc[mi][nt] = wmma_bf(af[mi], bf_, acc[mi][nt]);
    }
    __syncthreads();
  }
  const int rl = lane & 15, ho = (lane >> 4) << 3;
#pragma unroll
  for (int nt = 0; nt < 4; ++nt) {
    int n = n0 + wc * 64 + nt * 16 + rl;
    float bv = bias[n];
#pragma unroll
    for (int mi = 0; mi < MW; ++mi) {
#pragma unroll
      for (int i = 0; i < 8; ++i) {
        int m = m0 + (wr * MW + mi) * 16 + ho + i;
        float v = acc[mi][nt][i] + bv;
        if (HASC) v += Cin[(long)m * N + n];
        if (ACT == 1) v = v > 0.f ? v : 0.f;
        else if (ACT == 2) v = eluf(v);
        Out[(long)m * N + n] = v;
      }
    }
  }
}

// =====================================================================
// Dynamic adjacency: A_mid = elu(A_init.flat @ f1w + b1) (4096->256)
//   A_dyn = relu(A_mid @ f2w + b2) (256->4096); A_norm = row-normalize
// one block per branch
// =====================================================================
__global__ __launch_bounds__(256) void adj_k(
    const float* __restrict__ Ainit, const float* __restrict__ f1w,
    const float* __restrict__ f1b, const float* __restrict__ f2w,
    const float* __restrict__ f2b, float* __restrict__ Adyn,
    float* __restrict__ Anorm)
{
  __shared__ float sBuf[4096];
  __shared__ float sMid[256];
  __shared__ float sSum[64];
  const int br = blockIdx.x, t = threadIdx.x;
#pragma unroll
  for (int j = 0; j < 16; ++j) sBuf[t + j * 256] = Ainit[t + j * 256];
  __syncthreads();
  {
    const float* w = f1w + (long)br * 4096 * 256;
    float s = f1b[br * 256 + t];
    for (int i = 0; i < 4096; ++i) s += sBuf[i] * w[(long)i * 256 + t];
    sMid[t] = eluf(s);
  }
  __syncthreads();
  const float* w2 = f2w + (long)br * 256 * 4096;
  float outv[16];
#pragma unroll 1
  for (int j = 0; j < 16; ++j) {
    int n = t * 16 + j;
    float s = f2b[br * 4096 + n];
    for (int m = 0; m < 256; ++m) s += sMid[m] * w2[m * 4096 + n];
    outv[j] = s > 0.f ? s : 0.f;
  }
  __syncthreads();
#pragma unroll
  for (int j = 0; j < 16; ++j) {
    int n = t * 16 + j;
    sBuf[n] = outv[j];
    Adyn[br * 4096 + n] = outv[j];
  }
  __syncthreads();
  if (t < 64) {
    float s = 0.f;
    for (int j = 0; j < 64; ++j) s += sBuf[t * 64 + j];
    sSum[t] = s;
  }
  __syncthreads();
#pragma unroll
  for (int j = 0; j < 16; ++j) {
    int n = t * 16 + j;
    Anorm[br * 4096 + n] = sBuf[n] / (sSum[n >> 6] + 1e-6f);
  }
}

// =====================================================================
// G = elu(A_norm @ H + x), per (branch, batch); grid (4 col-chunks, B, 3)
// =====================================================================
__global__ __launch_bounds__(256) void mix_k(
    const float* __restrict__ Anorm, const float* __restrict__ H,
    const float* __restrict__ x, float* __restrict__ G)
{
  __shared__ __align__(16) __bf16 sAn[64 * 64];
  __shared__ __align__(16) __bf16 sBt[128 * 64];   // N-major chunk of H
  const int c0 = blockIdx.x * 128;
  const int b = blockIdx.y, br = blockIdx.z;
  const int t = threadIdx.x, wv = t >> 5, lane = t & 31;
  const float* An = Anorm + br * 4096;
  const long base = ((long)(br * BB + b)) * CCH * DD;
#pragma unroll
  for (int j = 0; j < 16; ++j) { int i = t + j * 256; sAn[i] = f2bf(An[i]); }
#pragma unroll 1
  for (int j = 0; j < 32; ++j) {
    int idx = t + j * 256;
    int n = idx >> 6, k = idx & 63;
    sBt[n * 64 + k] = f2bf(H[base + (long)k * DD + c0 + n]);
  }
  __syncthreads();
  const int rs = (wv & 3) * 16;
  const int ct0 = (wv >> 2) * 4;
  const v8f vz = {0.f, 0.f, 0.f, 0.f, 0.f, 0.f, 0.f, 0.f};
  v8f acc[4];
#pragma unroll
  for (int i = 0; i < 4; ++i) acc[i] = vz;
#pragma unroll
  for (int ks = 0; ks < 2; ++ks) {
    v16bf af = ldA16(sAn + rs * 64 + ks * 32, 64, lane);
#pragma unroll
    for (int nt = 0; nt < 4; ++nt) {
      v16bf bf_ = ldB16(sBt + (ct0 + nt) * 16 * 64 + ks * 32, 64, lane);
      acc[nt] = wmma_bf(af, bf_, acc[nt]);
    }
  }
  const int rl = lane & 15, ho = (lane >> 4) << 3;
#pragma unroll
  for (int nt = 0; nt < 4; ++nt) {
    int d = c0 + (ct0 + nt) * 16 + rl;
#pragma unroll
    for (int i = 0; i < 8; ++i) {
      int m = rs + ho + i;
      long off = base + (long)m * DD + d;
      G[off] = eluf(acc[nt][i] + x[off]);
    }
  }
}

// =====================================================================
// Fused per-(branch,batch) head: S=QK^T/8 -> softmax -> P@V -> A_dyn@(PV)
// writes directly into concatenated Z_local[b][c][br*64+v]
// =====================================================================
__global__ __launch_bounds__(256) void attn_k(
    const float* __restrict__ Q, const float* __restrict__ Km,
    const float* __restrict__ V, const float* __restrict__ Adyn,
    float* __restrict__ ZL)
{
  __shared__ __align__(16) unsigned char smem[49152];
  __bf16* sQ   = (__bf16*)(smem);            // 8KB
  __bf16* sKt  = (__bf16*)(smem + 8192);     // 8KB (K row-major == Bt for QK^T)
  __bf16* sVt  = (__bf16*)(smem + 16384);    // 8KB (V transposed)
  __bf16* sAd  = (__bf16*)(smem + 24576);    // 8KB
  __bf16* sP   = (__bf16*)(smem + 32768);    // 8KB
  __bf16* sPVt = (__bf16*)(smem + 40960);    // 8KB
  float*  sS   = (float*)(smem);             // 16KB, overlays sQ+sKt (dead then)

  const int b = blockIdx.x, br = blockIdx.y;
  const long qb = ((long)(br * BB + b)) * CCH * DKW;
  const int t = threadIdx.x, wv = t >> 5, lane = t & 31;
  const int rs = (wv & 3) * 16, ct0 = (wv >> 2) * 2;
  const int rl = lane & 15, ho = (lane >> 4) << 3;
  const v8f vz = {0.f, 0.f, 0.f, 0.f, 0.f, 0.f, 0.f, 0.f};

#pragma unroll
  for (int j = 0; j < 16; ++j) {
    int i = t + j * 256;
    int r = i >> 6, c = i & 63;
    sQ[i]           = f2bf(Q[qb + i]);
    sKt[i]          = f2bf(Km[qb + i]);
    sVt[c * 64 + r] = f2bf(V[qb + i]);
    sAd[i]          = f2bf(Adyn[br * 4096 + i]);
  }
  __syncthreads();

  // ---- S = Q @ K^T * 0.125 ----
  {
    v8f acc[2]; acc[0] = vz; acc[1] = vz;
#pragma unroll
    for (int ks = 0; ks < 2; ++ks) {
      v16bf af = ldA16(sQ + rs * 64 + ks * 32, 64, lane);
#pragma unroll
      for (int nt = 0; nt < 2; ++nt) {
        v16bf bf_ = ldB16(sKt + (ct0 + nt) * 16 * 64 + ks * 32, 64, lane);
        acc[nt] = wmma_bf(af, bf_, acc[nt]);
      }
    }
    __syncthreads();   // sQ/sKt reads done before sS overlay is written
#pragma unroll
    for (int nt = 0; nt < 2; ++nt)
#pragma unroll
      for (int i = 0; i < 8; ++i)
        sS[(rs + ho + i) * 64 + (ct0 + nt) * 16 + rl] = acc[nt][i] * 0.125f;
  }
  __syncthreads();

  // ---- softmax rows -> P (bf16) ----
  if (t < 64) {
    float mx = -1e30f;
    for (int j = 0; j < 64; ++j) mx = fmaxf(mx, sS[t * 64 + j]);
    float sum = 0.f;
    for (int j = 0; j < 64; ++j) { float e = __expf(sS[t * 64 + j] - mx); sS[t * 64 + j] = e; sum += e; }
    float inv = 1.f / sum;
    for (int j = 0; j < 64; ++j) sP[t * 64 + j] = f2bf(sS[t * 64 + j] * inv);
  }
  __syncthreads();

  // ---- PV = P @ V (store transposed for next stage) ----
  {
    v8f acc[2]; acc[0] = vz; acc[1] = vz;
#pragma unroll
    for (int ks = 0; ks < 2; ++ks) {
      v16bf af = ldA16(sP + rs * 64 + ks * 32, 64, lane);
#pragma unroll
      for (int nt = 0; nt < 2; ++nt) {
        v16bf bf_ = ldB16(sVt + (ct0 + nt) * 16 * 64 + ks * 32, 64, lane);
        acc[nt] = wmma_bf(af, bf_, acc[nt]);
      }
    }
#pragma unroll
    for (int nt = 0; nt < 2; ++nt)
#pragma unroll
      for (int i = 0; i < 8; ++i) {
        int m = rs + ho + i, v = (ct0 + nt) * 16 + rl;
        sPVt[v * 64 + m] = f2bf(acc[nt][i]);
      }
  }
  __syncthreads();

  // ---- Z = A_dyn @ PV -> Z_local ----
  {
    v8f acc[2]; acc[0] = vz; acc[1] = vz;
#pragma unroll
    for (int ks = 0; ks < 2; ++ks) {
      v16bf af = ldA16(sAd + rs * 64 + ks * 32, 64, lane);
#pragma unroll
      for (int nt = 0; nt < 2; ++nt) {
        v16bf bf_ = ldB16(sPVt + (ct0 + nt) * 16 * 64 + ks * 32, 64, lane);
        acc[nt] = wmma_bf(af, bf_, acc[nt]);
      }
    }
#pragma unroll
    for (int nt = 0; nt < 2; ++nt)
#pragma unroll
      for (int i = 0; i < 8; ++i) {
        int m = rs + ho + i, v = (ct0 + nt) * 16 + rl;
        ZL[((long)b * CCH + m) * DV3W + br * DKW + v] = acc[nt][i];
      }
  }
}

// channel weights: cw = softmax(mean over C of Z_local)
__global__ void cw_k(const float* __restrict__ ZL, float* __restrict__ CW)
{
  __shared__ float sm[DV3W];
  const int b = blockIdx.x, t = threadIdx.x;
  float s = 0.f;
  for (int c = 0; c < CCH; ++c) s += ZL[((long)b * CCH + c) * DV3W + t];
  sm[t] = s * (1.f / (float)CCH);
  __syncthreads();
  float mx = -1e30f;
  for (int j = 0; j < DV3W; ++j) mx = fmaxf(mx, sm[j]);
  float sum = 0.f;
  for (int j = 0; j < DV3W; ++j) sum += __expf(sm[j] - mx);
  CW[b * DV3W + t] = __expf(sm[t] - mx) / sum;
}

// Zg = (A_init @ Z_local) * cw, per batch
__global__ __launch_bounds__(256) void zglob_k(
    const float* __restrict__ Ainit, const float* __restrict__ ZL,
    const float* __restrict__ CW, float* __restrict__ Zg)
{
  __shared__ __align__(16) __bf16 sAi[64 * 64];
  __shared__ __align__(16) __bf16 sZt[DV3W * 64];
  const int b = blockIdx.x;
  const int t = threadIdx.x, wv = t >> 5, lane = t & 31;
#pragma unroll
  for (int j = 0; j < 16; ++j) { int i = t + j * 256; sAi[i] = f2bf(Ainit[i]); }
#pragma unroll 1
  for (int j = 0; j < 48; ++j) {
    int idx = t + j * 256;
    int n = idx >> 6, k = idx & 63;
    sZt[n * 64 + k] = f2bf(ZL[((long)b * CCH + k) * DV3W + n]);
  }
  __syncthreads();
  const int rs = (wv & 3) * 16, ct0 = (wv >> 2) * 6;
  const v8f vz = {0.f, 0.f, 0.f, 0.f, 0.f, 0.f, 0.f, 0.f};
  v8f acc[6];
#pragma unroll
  for (int i = 0; i < 6; ++i) acc[i] = vz;
#pragma unroll
  for (int ks = 0; ks < 2; ++ks) {
    v16bf af = ldA16(sAi + rs * 64 + ks * 32, 64, lane);
#pragma unroll
    for (int nt = 0; nt < 6; ++nt) {
      v16bf bf_ = ldB16(sZt + (ct0 + nt) * 16 * 64 + ks * 32, 64, lane);
      acc[nt] = wmma_bf(af, bf_, acc[nt]);
    }
  }
  const int rl = lane & 15, ho = (lane >> 4) << 3;
#pragma unroll
  for (int nt = 0; nt < 6; ++nt) {
    int d = (ct0 + nt) * 16 + rl;
#pragma unroll
    for (int i = 0; i < 8; ++i) {
      int m = rs + ho + i;
      Zg[((long)b * CCH + m) * DV3W + d] = acc[nt][i] * CW[b * DV3W + d];
    }
  }
}

// LayerNorm over last dim (192), one block per row
__global__ void ln_k(const float* __restrict__ Zg, const float* __restrict__ g,
                     const float* __restrict__ be, float* __restrict__ O)
{
  __shared__ float sm[DV3W];
  const long row = blockIdx.x;
  const int t = threadIdx.x;
  float v = Zg[row * DV3W + t];
  sm[t] = v;
  __syncthreads();
  float mu = 0.f;
  for (int j = 0; j < DV3W; ++j) mu += sm[j];
  mu *= (1.f / (float)DV3W);
  float var = 0.f;
  for (int j = 0; j < DV3W; ++j) { float d = sm[j] - mu; var += d * d; }
  var *= (1.f / (float)DV3W);
  O[row * DV3W + t] = (v - mu) * rsqrtf(var + 1e-5f) * g[t] + be[t];
}

// logits = hdn @ out2_w + out2_b  (512x128 @ 128x2)
__global__ void out2_k(const float* __restrict__ Hd, const float* __restrict__ W,
                       const float* __restrict__ bias, float* __restrict__ L)
{
  int i = blockIdx.x * blockDim.x + threadIdx.x;
  if (i >= BB * 2) return;
  int b = i >> 1, n = i & 1;
  float s = bias[n];
  for (int k = 0; k < HIDW; ++k) s += Hd[b * HIDW + k] * W[k * 2 + n];
  L[i] = s;
}

// =====================================================================
extern "C" void kernel_launch(void* const* d_in, const int* in_sizes, int n_in,
                              void* d_out, int out_size, void* d_ws, size_t ws_size,
                              hipStream_t stream)
{
  (void)in_sizes; (void)n_in; (void)out_size; (void)ws_size;
  const float* x      = (const float*)d_in[0];
  const float* A_init = (const float*)d_in[1];
  const float* f1w = (const float*)d_in[2];
  const float* f1b = (const float*)d_in[3];
  const float* f2w = (const float*)d_in[4];
  const float* f2b = (const float*)d_in[5];
  const float* q1w = (const float*)d_in[6];
  const float* q1b = (const float*)d_in[7];
  const float* q2w = (const float*)d_in[8];
  const float* q2b = (const float*)d_in[9];
  const float* wqw = (const float*)d_in[10];
  const float* wqb = (const float*)d_in[11];
  const float* wkw = (const float*)d_in[12];
  const float* wkb = (const float*)d_in[13];
  const float* wvw = (const float*)d_in[14];
  const float* wvb = (const float*)d_in[15];
  const float* lng = (const float*)d_in[16];
  const float* lnb = (const float*)d_in[17];
  const float* fc1w = (const float*)d_in[18];
  const float* fc1b = (const float*)d_in[19];
  const float* fc2w = (const float*)d_in[20];
  const float* fc2b = (const float*)d_in[21];
  const float* fm1w = (const float*)d_in[22];
  const float* fm1b = (const float*)d_in[23];
  const float* fm2w = (const float*)d_in[24];
  const float* fm2b = (const float*)d_in[25];
  const float* o1w = (const float*)d_in[26];
  const float* o1b = (const float*)d_in[27];
  const float* o2w = (const float*)d_in[28];
  const float* o2b = (const float*)d_in[29];

  float* ws = (float*)d_ws;
  const size_t SZ  = 3ull * BB * CCH * DD;        // 50,331,648 floats
  const size_t ZSZ = (size_t)BB * CCH * DV3W;     // 6,291,456
  const size_t QSZ = 3ull * BB * CCH * DKW;       // 6,291,456
  float* R0    = ws;            // T1 -> G -> {ZL, LN, F1, FC, M1}
  float* R1    = ws + SZ;       // H  -> {Q,K,V} -> Zg
  float* ADYN  = ws + 2 * SZ;
  float* ANORM = ADYN + 3 * 4096;
  float* CW    = ANORM + 3 * 4096;
  float* HDN   = CW + (size_t)BB * DV3W;

  float* Qb = R1;
  float* Kb = R1 + QSZ;
  float* Vb = R1 + 2 * QSZ;
  float* ZL = R0;
  float* ZG = R1;
  float* LN = R0 + ZSZ;
  float* F1 = R0 + 2 * ZSZ;
  float* FC = R0 + 3 * ZSZ;
  float* M1 = R0 + 4 * ZSZ;
  float* LOGITS = (float*)d_out;
  float* ZF     = (float*)d_out + (size_t)BB * 2;

  const int MBC = BB * CCH;                       // 32768

  // 0) dynamic adjacency
  adj_k<<<3, 256, 0, stream>>>(A_init, f1w, f1b, f2w, f2b, ADYN, ANORM);
  // 1) T1 = elu(x @ q1 + b)        (BN=128 path: 8 WMMA / kstep / wave)
  gemm_k<2, false, 128><<<dim3(DD / 128, MBC / 128, 3), 256, 0, stream>>>(
      x, q1w, q1b, nullptr, R0, MBC, DD, DD,
      (long)MBC * DD, (long)DD * DD, DD, (long)MBC * DD);
  // 2) H = T1 @ q2 + b
  gemm_k<0, false, 128><<<dim3(DD / 128, MBC / 128, 3), 256, 0, stream>>>(
      R0, q2w, q2b, nullptr, R1, MBC, DD, DD,
      (long)MBC * DD, (long)DD * DD, DD, (long)MBC * DD);
  // 3) G = elu(A_norm @ H + x)
  mix_k<<<dim3(4, BB, 3), 256, 0, stream>>>(ANORM, R1, x, R0);
  // 4) Q,K,V projections (N=64 -> BN=64 path)
  gemm_k<0, false, 64><<<dim3(1, MBC / 128, 3), 256, 0, stream>>>(
      R0, wqw, wqb, nullptr, Qb, MBC, DKW, DD,
      (long)MBC * DD, (long)DD * DKW, DKW, (long)MBC * DKW);
  gemm_k<0, false, 64><<<dim3(1, MBC / 128, 3), 256, 0, stream>>>(
      R0, wkw, wkb, nullptr, Kb, MBC, DKW, DD,
      (long)MBC * DD, (long)DD * DKW, DKW, (long)MBC * DKW);
  gemm_k<0, false, 64><<<dim3(1, MBC / 128, 3), 256, 0, stream>>>(
      R0, wvw, wvb, nullptr, Vb, MBC, DKW, DD,
      (long)MBC * DD, (long)DD * DKW, DKW, (long)MBC * DKW);
  // 5) fused attention + A_dyn mixing -> Z_local (concat over branches)
  attn_k<<<dim3(BB, 3), 256, 0, stream>>>(Qb, Kb, Vb, ADYN, ZL);
  // 6) channel weights, then Zg = (A_init @ Z_local) * cw
  cw_k<<<BB, DV3W, 0, stream>>>(ZL, CW);
  zglob_k<<<BB, 256, 0, stream>>>(A_init, ZL, CW, ZG);
  // 7) LayerNorm
  ln_k<<<MBC, DV3W, 0, stream>>>(ZG, lng, lnb, LN);
  // 8) fusion MLPs: Z_fc = relu(LN@fc1)@fc2 ; Z_fused = Z_fc + elu(Zg@fm1)@fm2
  gemm_k<1, false, 64><<<dim3(DV3W / 64, MBC / 128, 1), 256, 0, stream>>>(
      LN, fc1w, fc1b, nullptr, F1, MBC, DV3W, DV3W, 0, 0, 0, 0);
  gemm_k<0, false, 64><<<dim3(DV3W / 64, MBC / 128, 1), 256, 0, stream>>>(
      F1, fc2w, fc2b, nullptr, FC, MBC, DV3W, DV3W, 0, 0, 0, 0);
  gemm_k<2, false, 64><<<dim3(DV3W / 64, MBC / 128, 1), 256, 0, stream>>>(
      ZG, fm1w, fm1b, nullptr, M1, MBC, DV3W, DV3W, 0, 0, 0, 0);
  gemm_k<0, true, 64><<<dim3(DV3W / 64, MBC / 128, 1), 256, 0, stream>>>(
      M1, fm2w, fm2b, FC, ZF, MBC, DV3W, DV3W, 0, 0, 0, 0);
  // 9) output head (N=128 -> BN=128 path)
  gemm_k<1, false, 128><<<dim3(HIDW / 128, BB / 128, 1), 256, 0, stream>>>(
      ZF, o1w, o1b, nullptr, HDN, BB, HIDW, CCH * DV3W, 0, 0, 0, 0);
  out2_k<<<(BB * 2 + 255) / 256, 256, 0, stream>>>(HDN, o2w, o2b, LOGITS);
}
